// EvoformerNoMSABlock_23476291239970
// MI455X (gfx1250) — compile-verified, matrix operand order checked by reading, not run on hardware
//
#include <hip/hip_runtime.h>
#include <math.h>

#define DEV __device__ __forceinline__

typedef __attribute__((ext_vector_type(16))) __bf16 v16bf;
typedef __attribute__((ext_vector_type(8)))  float  v8f;

static const int R = 256;          // residues
static const int CM = 384;         // msa channels
static const int CZ = 128;         // pair channels
static const int RR = R * R;       // 65536 pair rows

// ---------- bf16 helpers (RNE, storage-only __bf16) ----------
DEV unsigned short f2bf_raw(float f) {
  unsigned int u = __float_as_uint(f);
  u += 0x7FFFu + ((u >> 16) & 1u);
  return (unsigned short)(u >> 16);
}
DEV __bf16 f2bf(float f) {
  unsigned short s = f2bf_raw(f);
  __bf16 r; __builtin_memcpy(&r, &s, 2); return r;
}
DEV float bf2f(unsigned short s) { return __uint_as_float((unsigned int)s << 16); }

// WMMA fragment index math for V_WMMA_F32_16X16X32_BF16 (wave32):
//  A (16x32): lane L -> m=L&15, g=L>>4; element e -> k = ((e&8)<<1) + 8g + (e&7)
//             => per lane: two contiguous 8-elem runs at k0+8g and k0+16+8g
//  B (32x16): lane L -> n=L&15, g=L>>4; element e -> same k formula
//  C/D (16x16 f32): lane L -> n=L&15, g=L>>4; vgpr v -> m = v + 8g
DEV int fragK(int e, int g) { return ((e & 8) << 1) + (g << 3) + (e & 7); }
DEV int rowswap(int r) { return ((r & 255) << 8) | (r >> 8); }   // (i,j)<->(j,i), R=256

DEV v8f wmma_bf16(v16bf a, v16bf b, v8f c) {
  return __builtin_amdgcn_wmma_f32_16x16x32_bf16(false, a, false, b, (short)0, c, false, false);
}

union V16U { v16bf v; uint4 q[2]; unsigned short us[16]; };

// two aligned 16B loads of a fragment's two 8-ushort runs (gap of 16 elements)
DEV v16bf load_frag_runs_u16(const unsigned short* __restrict__ p) {
  V16U u;
  u.q[0] = *(const uint4*)(p);
  u.q[1] = *(const uint4*)(p + 16);
  return u.v;
}
// 16 contiguous ushorts (pre-swizzled weight layout)
DEV v16bf load_frag_packed_u16(const unsigned short* __restrict__ p) {
  V16U u;
  u.q[0] = *(const uint4*)(p);
  u.q[1] = *(const uint4*)(p + 8);
  return u.v;
}

// vectorized load of one lane's 16 A-elements as f32: runs at p and p+16
DEV void load16(const float* __restrict__ p, float v[16]) {
  float4 x0 = *(const float4*)(p);
  float4 x1 = *(const float4*)(p + 4);
  float4 x2 = *(const float4*)(p + 16);
  float4 x3 = *(const float4*)(p + 20);
  v[0] = x0.x; v[1] = x0.y; v[2]  = x0.z; v[3]  = x0.w;
  v[4] = x1.x; v[5] = x1.y; v[6]  = x1.z; v[7]  = x1.w;
  v[8] = x2.x; v[9] = x2.y; v[10] = x2.z; v[11] = x2.w;
  v[12] = x3.x; v[13] = x3.y; v[14] = x3.z; v[15] = x3.w;
}

// =====================================================================
// Weight pre-swizzle: Wp[(kb*N + n)*32 + g*16 + e] = bf16(W[(kb*32+fragK(e,g))*N + n])
// =====================================================================
__global__ void packW_kernel(const float* __restrict__ W, unsigned short* __restrict__ Wp,
                             int K, int N) {
  int idx = blockIdx.x * 256 + threadIdx.x;
  if (idx >= K * N) return;
  int e = idx & 15;
  int g = (idx >> 4) & 1;
  int rest = idx >> 5;
  int n = rest % N;
  int kb = rest / N;
  int k = (kb << 5) + fragK(e, g);
  Wp[idx] = f2bf_raw(W[(size_t)k * N + n]);
}

// =====================================================================
// LayerNorm: one wave per row; f32-out and bf16-out variants.
// =====================================================================
__global__ void ln_kernel(const float* __restrict__ x, const float* __restrict__ gw,
                          const float* __restrict__ bw, float* __restrict__ y,
                          int rows, int C) {
  int wave = threadIdx.x >> 5, lane = threadIdx.x & 31;
  int row = blockIdx.x * 8 + wave;
  if (row >= rows) return;
  float vals[12];
  int nv = C >> 5;
  const float* xr = x + (size_t)row * C;
  float s = 0.f;
  for (int i = 0; i < nv; ++i) { vals[i] = xr[lane + i * 32]; s += vals[i]; }
  for (int off = 16; off; off >>= 1) s += __shfl_xor(s, off, 32);
  float mu = s / (float)C;
  float vs = 0.f;
  for (int i = 0; i < nv; ++i) { float d = vals[i] - mu; vs += d * d; }
  for (int off = 16; off; off >>= 1) vs += __shfl_xor(vs, off, 32);
  float inv = rsqrtf(vs / (float)C + 1e-5f);
  float* yr = y + (size_t)row * C;
  for (int i = 0; i < nv; ++i) {
    int c = lane + i * 32;
    yr[c] = gw[c] * (vals[i] - mu) * inv + bw[c];
  }
}

__global__ void ln_bf16_kernel(const float* __restrict__ x, const float* __restrict__ gw,
                               const float* __restrict__ bw, unsigned short* __restrict__ y,
                               int rows, int C) {
  int wave = threadIdx.x >> 5, lane = threadIdx.x & 31;
  int row = blockIdx.x * 8 + wave;
  if (row >= rows) return;
  float vals[12];
  int nv = C >> 5;
  const float* xr = x + (size_t)row * C;
  float s = 0.f;
  for (int i = 0; i < nv; ++i) { vals[i] = xr[lane + i * 32]; s += vals[i]; }
  for (int off = 16; off; off >>= 1) s += __shfl_xor(s, off, 32);
  float mu = s / (float)C;
  float vs = 0.f;
  for (int i = 0; i < nv; ++i) { float d = vals[i] - mu; vs += d * d; }
  for (int off = 16; off; off >>= 1) vs += __shfl_xor(vs, off, 32);
  float inv = rsqrtf(vs / (float)C + 1e-5f);
  unsigned short* yr = y + (size_t)row * C;
  for (int i = 0; i < nv; ++i) {
    int c = lane + i * 32;
    yr[c] = f2bf_raw(gw[c] * (vals[i] - mu) * inv + bw[c]);
  }
}

// =====================================================================
// Templated WMMA GEMM, NT-way register-blocked over N (one A fragment
// feeds NT wmmas).  B pre-swizzled bf16.  C = epi(alpha * A' @ B + bias)
//   AMODE 0: A'[r,k] = A[map(r)*K + k]    (ABF: A is bf16 row-major)
//   AMODE 1: A'[r,k] = Af[..] * A2f[..]   (f32 pair)
//   AMODE 2: A'[r,k] = Af[(r>>8)*32+k>>5] * A2f[(r&255)*32+(k&31)] (OPM outer;
//            right-row fragment K-invariant, hoisted)
//   ACT: 0 none, 1 relu, 2 sigmoid;  EMUL: *= G;  ERES: += Res;  WBF: bf16 out
// =====================================================================
template <int AMODE, int ABF, int ATRANS, int ACT, int EMUL, int CTRANS, int ERES, int WBF,
          int NT>
__global__ void gemm_kernel(const void* __restrict__ Av, const float* __restrict__ A2f,
                            const unsigned short* __restrict__ Bp, const float* __restrict__ bias,
                            const float* __restrict__ G, const float* __restrict__ Res,
                            void* __restrict__ Cv, int M, int N, int K, float alpha) {
  int wid = blockIdx.x * 8 + (threadIdx.x >> 5);
  int nwt = (N >> 4) / NT;               // wave-tiles per row-strip
  int total = (M >> 4) * nwt;
  if (wid >= total) return;
  int r0 = (wid / nwt) << 4;
  int n0 = (wid % nwt) * (NT << 4);

  int lane = threadIdx.x & 31;
  int mA = lane & 15, g = lane >> 4;
  int rowA = r0 + mA;
  int rrA = ATRANS ? rowswap(rowA) : rowA;
  const float* Af = (const float*)Av;
  const unsigned short* Ab = (const unsigned short*)Av;
  const unsigned short* bbase = Bp + ((size_t)(n0 + mA) << 5) + (g << 4);

  v8f acc[NT];
#pragma unroll
  for (int j = 0; j < NT; ++j) acc[j] = (v8f){0.f, 0.f, 0.f, 0.f, 0.f, 0.f, 0.f, 0.f};

  float vr[16];                       // OPM: right-row fragment, K-invariant
  if (AMODE == 2) load16(A2f + (rowA & 255) * 32 + (g << 3), vr);

  for (int k0 = 0; k0 < K; k0 += 32) {
    v16bf a;
    if (AMODE == 2) {
      float lv = Af[(rowA >> 8) * 32 + (k0 >> 5)];
#pragma unroll
      for (int e = 0; e < 16; ++e) a[e] = f2bf(lv * vr[e]);
    } else if (ABF) {
      a = load_frag_runs_u16(Ab + (size_t)rrA * K + k0 + (g << 3));
    } else {
      float va[16];
      load16(Af + (size_t)rrA * K + k0 + (g << 3), va);
      if (AMODE == 1) {
        float vb[16];
        load16(A2f + (size_t)rrA * K + k0 + (g << 3), vb);
#pragma unroll
        for (int e = 0; e < 16; ++e) va[e] *= vb[e];
      }
#pragma unroll
      for (int e = 0; e < 16; ++e) a[e] = f2bf(va[e]);
    }
    const unsigned short* bk = bbase + ((size_t)(k0 >> 5) * N << 5);
#pragma unroll
    for (int j = 0; j < NT; ++j) {
      v16bf b = load_frag_packed_u16(bk + j * 512);   // (16 cols)<<5 = 512
      acc[j] = wmma_bf16(a, b, acc[j]);
    }
  }

  int n = lane & 15;
#pragma unroll
  for (int j = 0; j < NT; ++j) {
#pragma unroll
    for (int v = 0; v < 8; ++v) {
      int m = v + (g << 3);
      int row = r0 + m, col = n0 + j * 16 + n;
      float x = acc[j][v] * alpha;
      if (bias) x += bias[col];
      if (ACT == 1) x = fmaxf(x, 0.f);
      else if (ACT == 2) x = 1.f / (1.f + __expf(-x));
      if (EMUL) x *= G[(size_t)row * N + col];
      int rc = CTRANS ? rowswap(row) : row;
      if (ERES) x += Res[(size_t)rc * N + col];
      if (WBF) ((unsigned short*)Cv)[(size_t)rc * N + col] = f2bf_raw(x);
      else     ((float*)Cv)[(size_t)rc * N + col] = x;
    }
  }
}

template <int AM, int ABF, int AT, int ACT, int EM, int CT, int ER, int WBF, int NT = 4>
static void launch_gemm(hipStream_t s, const void* A, const float* A2,
                        const unsigned short* Bp, const float* bias, const float* G,
                        const float* Res, void* C, int M, int N, int K, float alpha) {
  int wtiles = (M / 16) * (N / 16 / NT);
  gemm_kernel<AM, ABF, AT, ACT, EM, CT, ER, WBF, NT><<<(wtiles + 7) / 8, 256, 0, s>>>(
      A, A2, Bp, bias, G, Res, C, M, N, K, alpha);
}

// =====================================================================
// Pack bf16 pair tensor to channel-major bf16: Xp[c][i*256+k] = X[src(i,k)*CZ + c]
//   incoming=0: src = i*256+k   incoming=1: src = k*256+i
// =====================================================================
__global__ void pack_kernel(const unsigned short* __restrict__ X, unsigned short* __restrict__ Xp,
                            int incoming) {
  __shared__ unsigned short tile[32][33];
  int rb = (blockIdx.x >> 2) << 5;     // 2048 row-tiles
  int cb = (blockIdx.x & 3) << 5;      // 4 channel-tiles
  int tr = threadIdx.x >> 5;           // 0..7
  int tc = threadIdx.x & 31;
  for (int i = tr; i < 32; i += 8)
    tile[i][tc] = X[(size_t)(rb + i) * CZ + cb + tc];
  __syncthreads();
  int rsrc = rb + tc;
  int rd = incoming ? rowswap(rsrc) : rsrc;
  for (int i = tr; i < 32; i += 8)
    Xp[(size_t)(cb + i) * RR + rd] = tile[tc][i];
}

// =====================================================================
// Triangle einsum on packed bf16 channel-major operands, 16x64 per wave:
//   X[i,j,c] = sum_k Lp[c][i,k] * Rp[c][j,k]
// =====================================================================
__global__ void trimul_kernel(const unsigned short* __restrict__ Lp,
                              const unsigned short* __restrict__ Rp,
                              float* __restrict__ X) {
  int wid = blockIdx.x * 8 + (threadIdx.x >> 5);    // 128 c * 16 ti * 4 jw = 8192
  int c = wid >> 6;
  int t = wid & 63;
  int ti = (t >> 2) << 4;
  int jw = (t & 3) << 6;                            // 64-wide j span
  int lane = threadIdx.x & 31;
  int mA = lane & 15, g = lane >> 4;
  const unsigned short* lrow = Lp + (size_t)c * RR + (ti + mA) * 256 + (g << 3);
  const unsigned short* rbase = Rp + (size_t)c * RR + (jw + mA) * 256 + (g << 3);

  v8f acc[4];
#pragma unroll
  for (int j = 0; j < 4; ++j) acc[j] = (v8f){0.f, 0.f, 0.f, 0.f, 0.f, 0.f, 0.f, 0.f};

  for (int k0 = 0; k0 < 256; k0 += 32) {
    v16bf a = load_frag_runs_u16(lrow + k0);
#pragma unroll
    for (int j = 0; j < 4; ++j) {
      v16bf b = load_frag_runs_u16(rbase + j * 4096 + k0);   // 16 rows * 256
      acc[j] = wmma_bf16(a, b, acc[j]);
    }
  }
  int n = lane & 15;
#pragma unroll
  for (int j = 0; j < 4; ++j)
#pragma unroll
    for (int v = 0; v < 8; ++v) {
      int m = v + (g << 3);
      X[(size_t)((ti + m) * 256 + jw + j * 16 + n) * CZ + c] = acc[j][v];
    }
}

// =====================================================================
// Pair-bias projection: nb[h, r] = sum_c Lb[map(r), c] * fw[c, h]   (h=4)
// =====================================================================
__global__ void nb_kernel(const unsigned short* __restrict__ Lb, const float* __restrict__ fw,
                          float* __restrict__ NB, int trans) {
  int r = blockIdx.x * blockDim.x + threadIdx.x;
  if (r >= RR) return;
  int rr = trans ? rowswap(r) : r;
  float acc0 = 0.f, acc1 = 0.f, acc2 = 0.f, acc3 = 0.f;
  const unsigned short* lr = Lb + (size_t)rr * CZ;
  for (int c = 0; c < CZ; ++c) {
    float a = bf2f(lr[c]);
    acc0 += a * fw[c * 4 + 0];
    acc1 += a * fw[c * 4 + 1];
    acc2 += a * fw[c * 4 + 2];
    acc3 += a * fw[c * 4 + 3];
  }
  NB[0 * RR + r] = acc0; NB[1 * RR + r] = acc1;
  NB[2 * RR + r] = acc2; NB[3 * RR + r] = acc3;
}

// =====================================================================
// Fused triangle attention core. Block = (b, h, q-chunk of 32).
// Q/K/V bf16 global; K row-major LDS, V fragment-swizzled LDS; WMMA
// logits (+nb); in-LDS softmax with in-place f32->bf16 compaction;
// WMMA P@V. Static LDS = 64 KB.
// =====================================================================
__global__ void attn_kernel(const unsigned short* __restrict__ Qg,
                            const unsigned short* __restrict__ Kt,
                            const unsigned short* __restrict__ Vt,
                            const float* __restrict__ NB, float* __restrict__ O) {
  __shared__ unsigned short Ks[256 * 32];   // 16 KB, [key][d]
  __shared__ unsigned short Vs[256 * 32];   // 16 KB, swizzled [(kb*32+d)*32 + g*16 + e]
  __shared__ float Sgf[32 * 256];           // 32 KB logits / probs

  int b  = blockIdx.x >> 5;
  int h  = (blockIdx.x >> 3) & 3;
  int qc = blockIdx.x & 7;                  // 8 chunks of 32 q-rows
  int tid = threadIdx.x;
  int lane = tid & 31, wave = tid >> 5;
  int mA = lane & 15, g = lane >> 4;

  // stage K (b128 copies) and V (swizzled scatter), both already bf16
  for (int i = tid * 8; i < 256 * 32; i += 256 * 8) {
    int key = i >> 5, d0 = i & 31;
    size_t gi = (size_t)(b * 256 + key) * CZ + h * 32 + d0;
    *(uint4*)(Ks + i) = *(const uint4*)(Kt + gi);
    V16U vq; vq.q[0] = *(const uint4*)(Vt + gi);
    int kb = key >> 5, kl = key & 31;
    int gg = (kl >> 3) & 1;
    int ee = (kl & 7) | ((kl & 16) >> 1);
    int base = kb * 1024 + gg * 16 + ee;    // + d*32
#pragma unroll
    for (int d = 0; d < 8; ++d) Vs[base + (d0 + d) * 32] = vq.us[d];
  }
  __syncthreads();

  // logits: 32 q-rows x 256 keys = 2x16 tiles; wave w -> qt=w>>2, 4 k-tiles
  int qt = wave >> 2;
  int kt0 = (wave & 3) * 4;
  v16bf aQ = load_frag_runs_u16(
      Qg + (size_t)(b * 256 + qc * 32 + qt * 16 + mA) * CZ + h * 32 + (g << 3));

  for (int kt = kt0; kt < kt0 + 4; ++kt) {
    v8f acc = {0.f, 0.f, 0.f, 0.f, 0.f, 0.f, 0.f, 0.f};
    v16bf bb = load_frag_runs_u16(Ks + (kt * 16 + mA) * 32 + (g << 3));
    acc = wmma_bf16(aQ, bb, acc);
    int n = lane & 15;
#pragma unroll
    for (int v = 0; v < 8; ++v) {
      int rq = qt * 16 + v + (g << 3);
      int key = kt * 16 + n;
      int q = qc * 32 + rq;
      Sgf[rq * 256 + key] = acc[v] + NB[h * RR + q * 256 + key];
    }
  }
  __syncthreads();

  // softmax per row; compact to bf16 in place (ushort[k] over float[k], ascending-safe)
  if (tid < 32) {
    float* rowf = Sgf + tid * 256;
    float mx = -1e30f;
    for (int k = 0; k < 256; ++k) mx = fmaxf(mx, rowf[k]);
    float s = 0.f;
    for (int k = 0; k < 256; ++k) { float e = __expf(rowf[k] - mx); rowf[k] = e; s += e; }
    float inv = 1.f / s;
    unsigned short* rowu = (unsigned short*)rowf;
    for (int k = 0; k < 256; ++k) rowu[k] = f2bf_raw(rowf[k] * inv);
  }
  __syncthreads();

  // O = P @ V : 32x32 out = 2x2 tiles, waves 0..3
  if (wave < 4) {
    int oqt = wave >> 1, dt = wave & 1;
    v8f acc = {0.f, 0.f, 0.f, 0.f, 0.f, 0.f, 0.f, 0.f};
    const unsigned short* prow =
        (const unsigned short*)(Sgf + (oqt * 16 + mA) * 256) + (g << 3);
    const unsigned short* vbase = Vs + (dt * 16 + mA) * 32 + (g << 4);
    for (int k0 = 0; k0 < 256; k0 += 32) {
      v16bf a  = load_frag_runs_u16(prow + k0);
      v16bf bb = load_frag_packed_u16(vbase + (k0 >> 5) * 1024);
      acc = wmma_bf16(a, bb, acc);
    }
    int n = lane & 15;
#pragma unroll
    for (int v = 0; v < 8; ++v) {
      int q = qc * 32 + oqt * 16 + v + (g << 3);
      O[(size_t)(b * 256 + q) * CZ + h * 32 + dt * 16 + n] = acc[v];
    }
  }
}

// =====================================================================
// Fused transition: Out = Res + relu(Lb @ W1 + b1) @ W2 + b2
// Lb bf16; hidden bf16 in dynamic LDS; W1/W2 pre-swizzled.
// Phase 1 is 4-way N-blocked (A fragment shared by 4 wmmas).
// =====================================================================
__global__ void transition_kernel(const unsigned short* __restrict__ Lb,
                                  const unsigned short* __restrict__ W1p,
                                  const float* __restrict__ b1,
                                  const unsigned short* __restrict__ W2p,
                                  const float* __restrict__ b2, const float* __restrict__ Res,
                                  float* __restrict__ Out, int Cin, int Chid, int Cout) {
  extern __shared__ unsigned short hid[];   // [16][Chid] bf16
  int m0 = blockIdx.x * 16;
  int lane = threadIdx.x & 31, wave = threadIdx.x >> 5;
  int mA = lane & 15, g = lane >> 4;
  const unsigned short* arow = Lb + (size_t)(m0 + mA) * Cin + (g << 3);

  int nt1 = Chid >> 4;                      // multiple of 32 (512->32, 1536->96)
  for (int t4 = wave * 4; t4 < nt1; t4 += 32) {
    v8f acc[4];
#pragma unroll
    for (int j = 0; j < 4; ++j) acc[j] = (v8f){0.f, 0.f, 0.f, 0.f, 0.f, 0.f, 0.f, 0.f};
    const unsigned short* bbase = W1p + ((size_t)(t4 * 16 + mA) << 5) + (g << 4);
    for (int k0 = 0; k0 < Cin; k0 += 32) {
      v16bf a = load_frag_runs_u16(arow + k0);
      const unsigned short* bk = bbase + ((size_t)(k0 >> 5) * Chid << 5);
#pragma unroll
      for (int j = 0; j < 4; ++j) {
        v16bf b = load_frag_packed_u16(bk + j * 512);
        acc[j] = wmma_bf16(a, b, acc[j]);
      }
    }
    int n = lane & 15;
#pragma unroll
    for (int j = 0; j < 4; ++j)
#pragma unroll
      for (int v = 0; v < 8; ++v) {
        int m = v + (g << 3);
        int col = (t4 + j) * 16 + n;
        hid[m * Chid + col] = f2bf_raw(fmaxf(acc[j][v] + b1[col], 0.f));
      }
  }
  __syncthreads();

  int nt2 = Cout >> 4;
  for (int t = wave; t < nt2; t += 8) {
    v8f acc = {0.f, 0.f, 0.f, 0.f, 0.f, 0.f, 0.f, 0.f};
    const unsigned short* hrow = hid + mA * Chid + (g << 3);
    const unsigned short* bbase = W2p + ((size_t)(t * 16 + mA) << 5) + (g << 4);
    for (int k0 = 0; k0 < Chid; k0 += 32) {
      v16bf a = load_frag_runs_u16(hrow + k0);
      v16bf b = load_frag_packed_u16(bbase + ((size_t)(k0 >> 5) * Cout << 5));
      acc = wmma_bf16(a, b, acc);
    }
    int n = lane & 15;
#pragma unroll
    for (int v = 0; v < 8; ++v) {
      int m = v + (g << 3);
      int col = t * 16 + n;
      Out[(size_t)(m0 + m) * Cout + col] = acc[v] + b2[col] + Res[(size_t)(m0 + m) * Cout + col];
    }
  }
}

// =====================================================================
// Host orchestration
// =====================================================================
extern "C" void kernel_launch(void* const* d_in, const int* in_sizes, int n_in,
                              void* d_out, int out_size, void* d_ws, size_t ws_size,
                              hipStream_t stream) {
  (void)in_sizes; (void)n_in; (void)out_size; (void)ws_size;
  const float* msa_in  = (const float*)d_in[0];
  const float* pair_in = (const float*)d_in[1];
  auto IN = [&](int i) { return (const float*)d_in[i]; };
  // pytree (sorted-key) indices:
  // msa_col: attn{g_b,g_w,k_w,o_b,o_w,q_w,v_w}=2..8, ln{b,g}=9,10
  // msa_trans: l1{b,w}=11,12 l2{b,w}=13,14 ln{b,g}=15,16
  // opm: left{b,w}=17,18 ln{b,g}=19,20 out{b,w}=21,22 right{b,w}=23,24
  // pair_trans: l1{b,w}=25,26 l2{b,w}=27,28 ln{b,g}=29,30
  // tri_att_end base 31, tri_att_start base 41
  // tri_mul_in base 51, tri_mul_out base 67

  // ---- workspace layout ----
  float* ws = (float*)d_ws;
  const size_t PR = (size_t)RR * CZ;           // 8388608 floats
  float* Lbuf = ws;
  float* T1 = ws + PR;
  float* T2 = ws + 2 * PR;
  float* T3 = ws + 3 * PR;
  float* T4 = ws + 4 * PR;
  float* PA = ws + 5 * PR;
  float* NB = ws + 6 * PR;                     // 4*RR
  float* MSA_LN = NB + 4 * (size_t)RR;
  float* MSA_T1 = MSA_LN + (size_t)R * CM;
  float* MSA_T2 = MSA_T1 + (size_t)R * CM;
  float* M1     = MSA_T2 + (size_t)R * CM;
  unsigned short* wp = (unsigned short*)(M1 + (size_t)R * CM);  // packed weights (~4.5MB)

  float* out_msa  = (float*)d_out;             // R*CM
  float* out_pair = out_msa + (size_t)R * CM;  // RR*CZ

  unsigned short* Lb  = (unsigned short*)Lbuf;
  unsigned short* MLb = (unsigned short*)MSA_LN;

  auto lnb = [&](const float* x, const float* gw, const float* bw, unsigned short* y,
                 int rows, int C) {
    ln_bf16_kernel<<<(rows + 7) / 8, 256, 0, stream>>>(x, gw, bw, y, rows, C);
  };
  auto packW = [&](const float* W, int K, int N) {
    unsigned short* dst = wp;
    int total = K * N;
    packW_kernel<<<(total + 255) / 256, 256, 0, stream>>>(W, dst, K, N);
    wp += total;
    return (const unsigned short*)dst;
  };

  // ---- pre-swizzle all weights to bf16 (one-time; lives in L2) ----
  const unsigned short* Wv   = packW(IN(8),  CM, CM);
  const unsigned short* Wgm  = packW(IN(3),  CM, CM);
  const unsigned short* Wo   = packW(IN(6),  CM, CM);
  const unsigned short* Wmt1 = packW(IN(12), CM, 1536);
  const unsigned short* Wmt2 = packW(IN(14), 1536, CM);
  const unsigned short* Wopl = packW(IN(18), CM, 32);
  const unsigned short* Wopr = packW(IN(24), CM, 32);
  const unsigned short* Wopo = packW(IN(22), 1024, CZ);
  const unsigned short* Wpt1 = packW(IN(26), CZ, 512);
  const unsigned short* Wpt2 = packW(IN(28), 512, CZ);
  struct TM { const unsigned short *gate, *left, *lg, *out, *right, *rg; };
  auto packTM = [&](int o) {
    TM t;
    t.gate = packW(IN(o + 1), CZ, CZ);  t.left = packW(IN(o + 3), CZ, CZ);
    t.lg   = packW(IN(o + 5), CZ, CZ);  t.out  = packW(IN(o + 11), CZ, CZ);
    t.right= packW(IN(o + 13), CZ, CZ); t.rg   = packW(IN(o + 15), CZ, CZ);
    return t;
  };
  TM tmo = packTM(67), tmi = packTM(51);
  struct TA { const unsigned short *g, *k, *o, *q, *v; };
  auto packTA = [&](int o) {
    TA t;
    t.g = packW(IN(o + 1), CZ, CZ); t.k = packW(IN(o + 2), CZ, CZ);
    t.o = packW(IN(o + 4), CZ, CZ); t.q = packW(IN(o + 5), CZ, CZ);
    t.v = packW(IN(o + 6), CZ, CZ);
    return t;
  };
  TA tae = packTA(31), tas = packTA(41);

  // ================= MSA column attention (S=1 => softmax==1, o=v) =================
  lnb(msa_in, IN(10), IN(9), MLb, R, CM);
  launch_gemm<0,1,0,0,0,0,0,0>(stream, MLb, nullptr, Wv, nullptr, nullptr, nullptr,
                               MSA_T1, R, CM, CM, 1.f);                                 // v
  launch_gemm<0,1,0,2,0,0,0,0>(stream, MLb, nullptr, Wgm, IN(2), nullptr, nullptr,
                               MSA_T2, R, CM, CM, 1.f);                                 // g
  launch_gemm<1,0,0,0,0,0,1,0>(stream, MSA_T1, MSA_T2, Wo, IN(5), nullptr, msa_in,
                               M1, R, CM, CM, 1.f);                                     // +res

  // ================= MSA transition (fused) ========================================
  lnb(M1, IN(16), IN(15), MLb, R, CM);
  transition_kernel<<<R / 16, 256, (size_t)16 * 1536 * 2, stream>>>(
      MLb, Wmt1, IN(11), Wmt2, IN(13), M1, out_msa, CM, 1536, CM);

  // ================= Outer product mean -> pair (pair_in -> PA) ====================
  lnb(out_msa, IN(20), IN(19), MLb, R, CM);
  float* OPL  = MSA_T1;  // [256,32]
  float* OPRt = MSA_T2;
  launch_gemm<0,1,0,0,0,0,0,0,2>(stream, MLb, nullptr, Wopl, IN(17), nullptr, nullptr,
                                 OPL, R, 32, CM, 1.f);
  launch_gemm<0,1,0,0,0,0,0,0,2>(stream, MLb, nullptr, Wopr, IN(23), nullptr, nullptr,
                                 OPRt, R, 32, CM, 1.f);
  launch_gemm<2,0,0,0,0,0,1,0>(stream, OPL, OPRt, Wopo, IN(21), nullptr, pair_in,
                               PA, RR, CZ, 1024, 1.f);

  // ================= Triangle multiplications ======================================
  auto tri_mul = [&](int o, const TM& w, const float* Pin, float* Pout, int outgoing) {
    lnb(Pin, IN(o + 7), IN(o + 6), Lb, RR, CZ);                                         // ln_in
    unsigned short* LeftB  = (unsigned short*)T2;
    unsigned short* RightB = (unsigned short*)T3;
    launch_gemm<0,1,0,2,0,0,0,0>(stream, Lb, nullptr, w.lg, IN(o + 4), nullptr, nullptr,
                                 T1, RR, CZ, CZ, 1.f);                                  // sig(left_g)
    launch_gemm<0,1,0,0,1,0,0,1>(stream, Lb, nullptr, w.left, IN(o + 2), T1, nullptr,
                                 LeftB, RR, CZ, CZ, 1.f);                               // left*gate -> bf16
    launch_gemm<0,1,0,2,0,0,0,0>(stream, Lb, nullptr, w.rg, IN(o + 14), nullptr, nullptr,
                                 T1, RR, CZ, CZ, 1.f);                                  // sig(right_g)
    launch_gemm<0,1,0,0,1,0,0,1>(stream, Lb, nullptr, w.right, IN(o + 12), T1, nullptr,
                                 RightB, RR, CZ, CZ, 1.f);                              // right*gate -> bf16
    launch_gemm<0,1,0,2,0,0,0,0>(stream, Lb, nullptr, w.gate, IN(o + 0), nullptr, nullptr,
                                 T4, RR, CZ, CZ, 1.f);                                  // sig(gate)
    unsigned short* PkL = (unsigned short*)T1;
    unsigned short* PkR = (unsigned short*)Lbuf;
    pack_kernel<<<8192, 256, 0, stream>>>(LeftB, PkL, outgoing ? 0 : 1);
    pack_kernel<<<8192, 256, 0, stream>>>(RightB, PkR, outgoing ? 0 : 1);
    trimul_kernel<<<1024, 256, 0, stream>>>(PkL, PkR, T2);                              // einsum -> T2
    unsigned short* Xb = (unsigned short*)T3;
    lnb(T2, IN(o + 9), IN(o + 8), Xb, RR, CZ);                                          // ln_out -> bf16
    launch_gemm<0,1,0,0,1,0,1,0>(stream, Xb, nullptr, w.out, IN(o + 10), T4, Pin,
                                 Pout, RR, CZ, CZ, 1.f);                                // P + gate*out
  };
  tri_mul(67, tmo, PA, out_pair, 1);   // tri_mul_out (outgoing)
  tri_mul(51, tmi, out_pair, PA, 0);   // tri_mul_in  (incoming)

  // ================= Triangle attentions ===========================================
  auto tri_attn = [&](int o, const TA& w, const float* Pin, float* Pout, int trans) {
    lnb(Pin, IN(o + 9), IN(o + 8), Lb, RR, CZ);
    nb_kernel<<<RR / 256, 256, 0, stream>>>(Lb, IN(o + 7), NB, trans);
    const float qs = 0.17677669529663687f;   // 32^-0.5
    unsigned short* Qb = (unsigned short*)T1;
    unsigned short* Kb = (unsigned short*)T2;
    unsigned short* Vb = (unsigned short*)T3;
    unsigned short* OGb = (unsigned short*)T1;   // reuse after attn consumed Qb
    if (trans) {
      launch_gemm<0,1,1,0,0,0,0,1>(stream, Lb, nullptr, w.q, nullptr, nullptr, nullptr, Qb, RR, CZ, CZ, qs);
      launch_gemm<0,1,1,0,0,0,0,1>(stream, Lb, nullptr, w.k, nullptr, nullptr, nullptr, Kb, RR, CZ, CZ, 1.f);
      launch_gemm<0,1,1,0,0,0,0,1>(stream, Lb, nullptr, w.v, nullptr, nullptr, nullptr, Vb, RR, CZ, CZ, 1.f);
      attn_kernel<<<8192, 256, 0, stream>>>(Qb, Kb, Vb, NB, T4);
      launch_gemm<0,1,1,2,1,0,0,1>(stream, Lb, nullptr, w.g, IN(o + 0), T4, nullptr, OGb, RR, CZ, CZ, 1.f);
      launch_gemm<0,1,0,0,0,1,1,0>(stream, OGb, nullptr, w.o, IN(o + 3), nullptr, Pin, Pout, RR, CZ, CZ, 1.f);
    } else {
      launch_gemm<0,1,0,0,0,0,0,1>(stream, Lb, nullptr, w.q, nullptr, nullptr, nullptr, Qb, RR, CZ, CZ, qs);
      launch_gemm<0,1,0,0,0,0,0,1>(stream, Lb, nullptr, w.k, nullptr, nullptr, nullptr, Kb, RR, CZ, CZ, 1.f);
      launch_gemm<0,1,0,0,0,0,0,1>(stream, Lb, nullptr, w.v, nullptr, nullptr, nullptr, Vb, RR, CZ, CZ, 1.f);
      attn_kernel<<<8192, 256, 0, stream>>>(Qb, Kb, Vb, NB, T4);
      launch_gemm<0,1,0,2,1,0,0,1>(stream, Lb, nullptr, w.g, IN(o + 0), T4, nullptr, OGb, RR, CZ, CZ, 1.f);
      launch_gemm<0,1,0,0,0,0,1,0>(stream, OGb, nullptr, w.o, IN(o + 3), nullptr, Pin, Pout, RR, CZ, CZ, 1.f);
    }
  };
  tri_attn(41, tas, PA, out_pair, 0);  // tri_att_start
  tri_attn(31, tae, out_pair, PA, 1);  // tri_att_end

  // ================= Pair transition (fused) -> final pair in d_out ================
  lnb(PA, IN(30), IN(29), Lb, RR, CZ);
  transition_kernel<<<RR / 16, 256, (size_t)16 * 512 * 2, stream>>>(
      Lb, Wpt1, IN(25), Wpt2, IN(27), PA, out_pair, CZ, 512, CZ);
}
